// Attention_13134009991675
// MI455X (gfx1250) — matrix operand order repaired and checked by reading, not compiled, not run on hardware
//
#include <hip/hip_runtime.h>
#include <hip/hip_bf16.h>
#include <math.h>

#define B_ 32
#define L_ 1024
#define D_ 2048
#define A_ 2048
#define NEG_SLOPE 0.01f
#define ROWS 64
#define RSTRIDE 2056  // bf16 elems per LDS row: 2048 + 8 pad (16B) -> conflict-free b128

typedef __attribute__((ext_vector_type(16))) __bf16 v16bf;
typedef __attribute__((ext_vector_type(8)))  float  v8f;

union Frag { v16bf v; uint4 u[2]; };

__device__ __forceinline__ float lrelu(float x){ return x >= 0.f ? x : NEG_SLOPE * x; }

// Branchless tanh: 1 - 2/(e^{2x}+1). Saturates to +/-1 for large |x| (e->inf / e->0),
// no EXEC divergence; v_exp_f32 is a TRANS op that co-executes with VALU.
__device__ __forceinline__ float fast_tanh(float x){
  float e = __expf(2.f * x);
  return 1.f - __fdividef(2.f, e + 1.f);
}

// fp32 -> bf16 round-to-nearest-even (bit trick)
__device__ __forceinline__ unsigned short f2bf(float f){
  unsigned u  = __float_as_uint(f);
  unsigned rb = ((u >> 16) & 1u) + 0x7FFFu;
  return (unsigned short)((u + rb) >> 16);
}
__device__ __forceinline__ unsigned pack2bf(float lo, float hi){
  return (unsigned)f2bf(lo) | ((unsigned)f2bf(hi) << 16);
}

// ---------------- Kernel 1: Wv fp32 -> bf16 (row-major (A, D)) ----------------
__global__ void k_cvt_wv(const float* __restrict__ wv, unsigned short* __restrict__ out){
  size_t i = ((size_t)blockIdx.x * blockDim.x + threadIdx.x) * 4;
  float4 f = *(const float4*)(wv + i);
  uint2 p; p.x = pack2bf(f.x, f.y); p.y = pack2bf(f.z, f.w);
  *(uint2*)(out + i) = p;
}

// ---------------- Kernel 2: qe = lrelu(query @ Wq^T + bq), (B, A) fp32 -------
__global__ void k_qe(const float* __restrict__ query, const float* __restrict__ Wq,
                     const float* __restrict__ bq, float* __restrict__ qe){
  __shared__ float sq[D_];
  const int b = blockIdx.y;
  const int a = blockIdx.x * 128 + threadIdx.x;
  for (int d = threadIdx.x * 4; d < D_; d += 128 * 4)
    *(float4*)&sq[d] = *(const float4*)&query[(size_t)b * D_ + d];
  __syncthreads();
  const float4* wr = (const float4*)(Wq + (size_t)a * D_);
  float acc = 0.f;
  #pragma unroll 4
  for (int d4 = 0; d4 < D_ / 4; ++d4){
    float4 w = wr[d4];
    float4 q = *(const float4*)&sq[d4 * 4];
    acc = fmaf(w.x, q.x, fmaf(w.y, q.y, fmaf(w.z, q.z, fmaf(w.w, q.w, acc))));
  }
  qe[(size_t)b * A_ + a] = lrelu(acc + bq[a]);
}

// ---------------- Kernel 3: fused GEMM (bf16 WMMA) + tanh/Wo reduction -------
__global__ void __launch_bounds__(256, 1) k_logits(
    const float* __restrict__ values, const unsigned short* __restrict__ wvb,
    const float* __restrict__ bv, const float* __restrict__ qe,
    const float* __restrict__ Wo, const float* __restrict__ bo,
    const int* __restrict__ mask, float* __restrict__ logits)
{
  extern __shared__ __align__(16) unsigned char smem[];
  __shared__ float s_logit[ROWS];
  const int tid  = threadIdx.x;
  const int row0 = blockIdx.x * ROWS;   // flat (b,l) row base; 64 | 1024 so b is constant
  const int b    = row0 / L_;

  // Stage 64 values rows into LDS as bf16 (padded rows).
  for (int i = tid * 4; i < ROWS * D_; i += 256 * 4){
    int r = i >> 11;          // / 2048
    int d = i & (D_ - 1);
    float4 f = *(const float4*)(values + ((size_t)(row0 + r)) * D_ + d);
    uint2 p; p.x = pack2bf(f.x, f.y); p.y = pack2bf(f.z, f.w);
    *(uint2*)(smem + (size_t)r * RSTRIDE * 2 + (size_t)d * 2) = p;
  }
  if (tid < ROWS) s_logit[tid] = 0.f;
  __syncthreads();

  const int lane  = tid & 31;
  const int wave  = tid >> 5;
  const int rlane = lane & 15;
  const int half  = lane >> 4;
  const int rt    = wave & 3;    // row tile 0..3 (16 rows each)
  const int ahalf = wave >> 2;   // a-tile interleave 0/1

  const uint4* arow = (const uint4*)(smem + (size_t)(rt * 16 + rlane) * RSTRIDE * 2);
  const float* qrow = qe + (size_t)b * A_;

  float lacc[8] = {0.f,0.f,0.f,0.f,0.f,0.f,0.f,0.f};

  for (int at = ahalf; at < A_ / 16; at += 2){
    const int a = at * 16 + rlane;            // this lane's output column
    const unsigned short* bptr = wvb + (size_t)a * D_ + half * 16;
    __builtin_prefetch(bptr + 32 * D_, 0, 1); // next a-tile for this wave
    // Issue epilogue scalars now; latency hides under the 64-deep WMMA loop.
    const float bvv = bv[a];
    const float qev = qrow[a];
    const float wov = Wo[a];

    // Two independent accumulator chains so consecutive WMMAs have no RAW dep.
    v8f c0 = {0.f,0.f,0.f,0.f,0.f,0.f,0.f,0.f};
    v8f c1 = {0.f,0.f,0.f,0.f,0.f,0.f,0.f,0.f};
    #pragma unroll 2
    for (int k0 = 0; k0 < D_; k0 += 64){
      const int a4 = k0 >> 3;                 // uint4 index of k0 within the row
      const uint4* bp = (const uint4*)(bptr + k0);
      Frag fa0, fb0, fa1, fb1;
      // A 16x32 bf16: lanes 0-15 K=[0..16), lanes 16-31 K=[16..32)
      fa0.u[0] = arow[a4 + 2 * half];
      fa0.u[1] = arow[a4 + 1 + 2 * half];
      fb0.u[0] = bp[0];
      fb0.u[1] = bp[1];
      fa1.u[0] = arow[a4 + 4 + 2 * half];
      fa1.u[1] = arow[a4 + 5 + 2 * half];
      fb1.u[0] = bp[4];
      fb1.u[1] = bp[5];
      c0 = __builtin_amdgcn_wmma_f32_16x16x32_bf16(false, fa0.v, false, fb0.v,
                                                   (short)0, c0, false, false);
      c1 = __builtin_amdgcn_wmma_f32_16x16x32_bf16(false, fa1.v, false, fb1.v,
                                                   (short)0, c1, false, false);
    }
    v8f c = c0 + c1;

    // Fused epilogue: lrelu(x+bv) + qe, tanh, *Wo, accumulate per-row logit.
    #pragma unroll
    for (int i = 0; i < 8; ++i){
      float x = lrelu(c[i] + bvv) + qev;
      lacc[i] = fmaf(fast_tanh(x), wov, lacc[i]);
    }
  }

  // C/D layout: VGPR i -> row (i + 8*half), lane%16 -> column. Reduce 16 columns.
  #pragma unroll
  for (int i = 0; i < 8; ++i){
    float v = lacc[i];
    v += __shfl_xor(v, 1);
    v += __shfl_xor(v, 2);
    v += __shfl_xor(v, 4);
    v += __shfl_xor(v, 8);
    if (rlane == 0)
      atomicAdd(&s_logit[rt * 16 + i + half * 8], v);
  }
  __syncthreads();

  if (tid < ROWS){
    int row = row0 + tid;
    float lg = lrelu(s_logit[tid] + bo[0]);
    logits[row] = (mask[row] == 0) ? -1e-9f : lg;  // reference's exact fill value
  }
}

// ---------------- Kernel 4: softmax over L per batch -------------------------
__global__ void k_softmax(const float* __restrict__ logits, float* __restrict__ probs){
  __shared__ float sred[256];
  const int b = blockIdx.x, tid = threadIdx.x;
  const float* lg = logits + (size_t)b * L_;
  float x0 = lg[tid], x1 = lg[tid + 256], x2 = lg[tid + 512], x3 = lg[tid + 768];
  float m = fmaxf(fmaxf(x0, x1), fmaxf(x2, x3));
  sred[tid] = m; __syncthreads();
  for (int s = 128; s > 0; s >>= 1){
    if (tid < s) sred[tid] = fmaxf(sred[tid], sred[tid + s]);
    __syncthreads();
  }
  const float mx = sred[0];
  __syncthreads();
  float e0 = __expf(x0 - mx), e1 = __expf(x1 - mx);
  float e2 = __expf(x2 - mx), e3 = __expf(x3 - mx);
  sred[tid] = e0 + e1 + e2 + e3; __syncthreads();
  for (int s = 128; s > 0; s >>= 1){
    if (tid < s) sred[tid] += sred[tid + s];
    __syncthreads();
  }
  const float inv = 1.f / sred[0];
  float* pr = probs + (size_t)b * L_;
  pr[tid] = e0 * inv; pr[tid + 256] = e1 * inv;
  pr[tid + 512] = e2 * inv; pr[tid + 768] = e3 * inv;
}

// ---------------- Kernel 5: out[b,d] = sum_l prob[b,l] * values[b,l,d] -------
__global__ void k_out(const float* __restrict__ probs, const float* __restrict__ values,
                      float* __restrict__ out){
  __shared__ float sp[L_];
  const int b = blockIdx.y;
  const int d = blockIdx.x * 256 + threadIdx.x;
  for (int l = threadIdx.x; l < L_; l += 256) sp[l] = probs[(size_t)b * L_ + l];
  __syncthreads();
  const float* vc = values + (size_t)b * L_ * D_ + d;
  float acc = 0.f;
  #pragma unroll 8
  for (int l = 0; l < L_; ++l)
    acc = fmaf(sp[l], vc[(size_t)l * D_], acc);
  out[(size_t)b * D_ + d] = acc;
}

// ---------------- Host launcher ----------------------------------------------
extern "C" void kernel_launch(void* const* d_in, const int* in_sizes, int n_in,
                              void* d_out, int out_size, void* d_ws, size_t ws_size,
                              hipStream_t stream){
  const float* query  = (const float*)d_in[0];
  const float* values = (const float*)d_in[1];
  const int*   mask   = (const int*)  d_in[2];
  const float* Wq     = (const float*)d_in[3];
  const float* bq     = (const float*)d_in[4];
  const float* Wv     = (const float*)d_in[5];
  const float* bv     = (const float*)d_in[6];
  const float* Wo     = (const float*)d_in[7];
  const float* bo     = (const float*)d_in[8];
  float* out = (float*)d_out;

  char* ws = (char*)d_ws;
  float*          qe     = (float*)(ws);                               // 32*2048*4   = 256 KB
  float*          logits = (float*)(ws + 262144);                      // 32768*4     = 128 KB
  float*          probs  = (float*)(ws + 262144 + 131072);             // 32768*4     = 128 KB
  unsigned short* wvb    = (unsigned short*)(ws + 262144 + 2*131072);  // 2048*2048*2 = 8 MB

  // 1) Wv -> bf16 (stays hot in 192 MB L2 for the whole GEMM)
  k_cvt_wv<<<(D_ * A_) / (256 * 4), 256, 0, stream>>>(Wv, wvb);
  // 2) query projection
  k_qe<<<dim3(A_ / 128, B_), 128, 0, stream>>>(query, Wq, bq, qe);
  // 3) fused big GEMM + epilogue -> logits (dynamic LDS: 64 padded bf16 rows)
  const size_t lds_bytes = (size_t)ROWS * RSTRIDE * 2;  // 263,168 B (< 320 KB WGP LDS)
  k_logits<<<(B_ * L_) / ROWS, 256, lds_bytes, stream>>>(values, wvb, bv, qe, Wo, bo,
                                                         mask, logits);
  // 4) softmax per batch
  k_softmax<<<B_, 256, 0, stream>>>(logits, probs);
  // 5) prob-weighted sum of values
  k_out<<<dim3(D_ / 256, B_), 256, 0, stream>>>(probs, values, out);
}
